// MIL_76295799046843
// MI455X (gfx1250) — matrix operand order repaired and checked by reading, not compiled, not run on hardware
//
#include <hip/hip_runtime.h>
#include <math.h>

// Problem dims
#define Bn   32
#define Tn   2048
#define Din  1024
#define H1n  512
#define H2n  32

#define ROWS 64            // rows (t-positions) per block
#define NWAVE 16           // waves per block (512 threads)
#define HSTRC 5            // Hs row stride in 16B chunks (40 bf16 = 80 B, conflict-free)

typedef __attribute__((ext_vector_type(16))) __bf16 v16bf;
typedef __attribute__((ext_vector_type(8)))  float  v8f;

union BF16Frag { uint4 q[2]; v16bf v; };

static __device__ inline unsigned short bf16_bits(float x) {
    union { float f; unsigned u; } c; c.f = x;
    unsigned u = c.u;
    unsigned r = (u + 0x7FFFu + ((u >> 16) & 1u)) >> 16;   // RNE fp32->bf16
    return (unsigned short)r;
}

// Shared memory: A-stage (128 KB, 64 rows x 1024 bf16 in XOR-swizzled 16B
// chunks) overlapped with layer-2 buffers (88 KB). ISA: up to 320 KB LDS/WG.
union SMem {
    uint4 As4[ROWS * 128];                     // [row][chunk^swz], 128 chunks/row
    struct {
        uint4 Hs4[NWAVE][ROWS * HSTRC];        // per-wave h tile (64 x 32 bf16, padded)
        float Acc[ROWS * H2n];                 // 64 x 32 f32 cross-wave partials
    } p2;
};

// ---------------------------------------------------------------------------
// Kernel 0: weight prep. W1t[n][k] = bf16(W1[k][n]), W2t[n][k] = bf16(W2[k][n])
// ---------------------------------------------------------------------------
__global__ __launch_bounds__(256) void prep_kernel(
    const float* __restrict__ W1, const float* __restrict__ W2,
    unsigned short* __restrict__ W1t, unsigned short* __restrict__ W2t)
{
    int idx = blockIdx.x * 256 + threadIdx.x;
    if (idx < H1n * Din) {
        int n = idx >> 10;
        int k = idx & 1023;
        W1t[idx] = bf16_bits(W1[k * H1n + n]);
    }
    if (idx < H2n * H1n) {
        int n = idx >> 9;
        int k = idx & 511;
        W2t[idx] = bf16_bits(W2[k * H2n + n]);
    }
}

// ---------------------------------------------------------------------------
// Kernel 1: fused 3-layer MLP + sigmoid -> logits[B*T].
// Block = 512 threads (16 wave32), 64 rows. Wave w owns H1 columns
// [w*32, w*32+32); register-blocks 4 row-tiles x 2 N-tiles (8 WMMA accums);
// each B fragment is reused 4x.
// ---------------------------------------------------------------------------
__global__ __launch_bounds__(512) void mlp_kernel(
    const float* __restrict__ avf,
    const unsigned short* __restrict__ W1t,
    const unsigned short* __restrict__ W2t,
    const float* __restrict__ b1, const float* __restrict__ b2,
    const float* __restrict__ W3, const float* __restrict__ b3,
    float* __restrict__ logits)
{
    __shared__ SMem sm;

    const int tid  = threadIdx.x;
    const int lane = tid & 31;
    const int wave = tid >> 5;
    const long r0  = (long)blockIdx.x * ROWS;

    // ---- stage A: ROWS x 1024 f32 -> bf16, XOR-swizzled 16B chunks in LDS
    {
        const float4* src = (const float4*)(avf + r0 * Din);
        #pragma unroll 4
        for (int i = 0; i < 32; ++i) {
            int j = tid + i * 512;                 // float4 index, < 16384
            float4 f = src[j];
            uint2 p;
            p.x = (unsigned)bf16_bits(f.x) | ((unsigned)bf16_bits(f.y) << 16);
            p.y = (unsigned)bf16_bits(f.z) | ((unsigned)bf16_bits(f.w) << 16);
            int row   = j >> 8;                    // 256 float4 per row
            int o     = (j & 255) << 3;            // byte offset within row (bf16 space)
            int chunk = (o >> 4) ^ (row & 15);
            int half  = j & 1;
            ((uint2*)&sm.As4[(row << 7) + chunk])[half] = p;
        }
    }
    __syncthreads();

    const int nl    = lane & 15;                   // N within a 16-wide tile
    const int klo   = (lane < 16) ? 0 : 8;         // A-frag K sub-chunk (elements)
    const int kboff = (lane < 16) ? 0 : 16;        // B-frag K half (elements)
    const int mb    = (lane >= 16) ? 8 : 0;        // C-frag row base
    const int sw    = nl;                          // LDS swizzle key (== row & 15)

    // ---- layer 1: K = 1024, 4 row-tiles x 2 N-tiles per wave
    v8f a00 = {0,0,0,0,0,0,0,0}, a01 = {0,0,0,0,0,0,0,0};
    v8f a10 = {0,0,0,0,0,0,0,0}, a11 = {0,0,0,0,0,0,0,0};
    v8f a20 = {0,0,0,0,0,0,0,0}, a21 = {0,0,0,0,0,0,0,0};
    v8f a30 = {0,0,0,0,0,0,0,0}, a31 = {0,0,0,0,0,0,0,0};

    for (int k = 0; k < Din; k += 32) {
        int cc  = (k + klo) >> 3;                  // 16B-chunk index within row
        int c0  = cc ^ sw;
        int c1  = (cc + 2) ^ sw;
        BF16Frag fa0, fa1, fa2, fa3;
        {
            const int b0 = (nl +  0) << 7;
            const int b1r = (nl + 16) << 7;
            const int b2r = (nl + 32) << 7;
            const int b3r = (nl + 48) << 7;
            fa0.q[0] = sm.As4[b0  + c0]; fa0.q[1] = sm.As4[b0  + c1];
            fa1.q[0] = sm.As4[b1r + c0]; fa1.q[1] = sm.As4[b1r + c1];
            fa2.q[0] = sm.As4[b2r + c0]; fa2.q[1] = sm.As4[b2r + c1];
            fa3.q[0] = sm.As4[b3r + c0]; fa3.q[1] = sm.As4[b3r + c1];
        }
        BF16Frag fb0, fb1;
        {
            const uint4* g;
            int nbase = wave * 32 + nl;
            g = (const uint4*)(W1t + (size_t)(nbase +  0) * Din + k + klo);
            fb0.q[0] = g[0]; fb0.q[1] = g[2];
            g = (const uint4*)(W1t + (size_t)(nbase + 16) * Din + k + klo);
            fb1.q[0] = g[0]; fb1.q[1] = g[2];
        }
        a00 = __builtin_amdgcn_wmma_f32_16x16x32_bf16(false, fa0.v, false, fb0.v, (short)0, a00, false, false);
        a01 = __builtin_amdgcn_wmma_f32_16x16x32_bf16(false, fa0.v, false, fb1.v, (short)0, a01, false, false);
        a10 = __builtin_amdgcn_wmma_f32_16x16x32_bf16(false, fa1.v, false, fb0.v, (short)0, a10, false, false);
        a11 = __builtin_amdgcn_wmma_f32_16x16x32_bf16(false, fa1.v, false, fb1.v, (short)0, a11, false, false);
        a20 = __builtin_amdgcn_wmma_f32_16x16x32_bf16(false, fa2.v, false, fb0.v, (short)0, a20, false, false);
        a21 = __builtin_amdgcn_wmma_f32_16x16x32_bf16(false, fa2.v, false, fb1.v, (short)0, a21, false, false);
        a30 = __builtin_amdgcn_wmma_f32_16x16x32_bf16(false, fa3.v, false, fb0.v, (short)0, a30, false, false);
        a31 = __builtin_amdgcn_wmma_f32_16x16x32_bf16(false, fa3.v, false, fb1.v, (short)0, a31, false, false);
    }

    // everyone done reading As -> reuse LDS for layer-2 buffers
    __syncthreads();

    // zero cross-wave accumulator (ROWS*32 = 2048 floats)
    #pragma unroll
    for (int i = 0; i < 4; ++i) sm.p2.Acc[tid + i * 512] = 0.0f;

    // ---- bias + ReLU, spill h tiles (4 x 16x32 bf16) into per-wave buffer
    {
        float bv0 = b1[wave * 32 + nl +  0];
        float bv1 = b1[wave * 32 + nl + 16];
        unsigned short* hsp = (unsigned short*)sm.p2.Hs4[wave];
        #pragma unroll
        for (int j = 0; j < 8; ++j) {
            int rA = (mb + j) * (HSTRC * 8);           // row stride = 40 elements
            hsp[rA + nl +  0]                    = bf16_bits(fmaxf(a00[j] + bv0, 0.0f));
            hsp[rA + nl + 16]                    = bf16_bits(fmaxf(a01[j] + bv1, 0.0f));
            hsp[rA + 16 * 40 + nl +  0]          = bf16_bits(fmaxf(a10[j] + bv0, 0.0f));
            hsp[rA + 16 * 40 + nl + 16]          = bf16_bits(fmaxf(a11[j] + bv1, 0.0f));
            hsp[rA + 32 * 40 + nl +  0]          = bf16_bits(fmaxf(a20[j] + bv0, 0.0f));
            hsp[rA + 32 * 40 + nl + 16]          = bf16_bits(fmaxf(a21[j] + bv1, 0.0f));
            hsp[rA + 48 * 40 + nl +  0]          = bf16_bits(fmaxf(a30[j] + bv0, 0.0f));
            hsp[rA + 48 * 40 + nl + 16]          = bf16_bits(fmaxf(a31[j] + bv1, 0.0f));
        }
    }
    __syncthreads();   // Acc zeroed by all waves before atomics

    // ---- layer 2: h(4 x 16x32) x W2[w*32 : w*32+32, 0:32] -> partial 64x32
    {
        int kg = wave * 32 + klo;
        BF16Frag w20, w21;
        {
            const uint4* g;
            g = (const uint4*)(W2t + (size_t)(nl +  0) * H1n + kg);
            w20.q[0] = g[0]; w20.q[1] = g[2];
            g = (const uint4*)(W2t + (size_t)(nl + 16) * H1n + kg);
            w21.q[0] = g[0]; w21.q[1] = g[2];
        }
        const int kc = klo >> 3;                   // A2-frag chunk offset (0 or 1)
        #pragma unroll
        for (int rt = 0; rt < 4; ++rt) {
            BF16Frag h;
            int cbase = (rt * 16 + nl) * HSTRC + kc;
            h.q[0] = sm.p2.Hs4[wave][cbase];
            h.q[1] = sm.p2.Hs4[wave][cbase + 2];
            v8f pa = {0,0,0,0,0,0,0,0};
            v8f pb = {0,0,0,0,0,0,0,0};
            pa = __builtin_amdgcn_wmma_f32_16x16x32_bf16(false, h.v, false, w20.v, (short)0, pa, false, false);
            pb = __builtin_amdgcn_wmma_f32_16x16x32_bf16(false, h.v, false, w21.v, (short)0, pb, false, false);
            #pragma unroll
            for (int j = 0; j < 8; ++j) {
                atomicAdd(&sm.p2.Acc[(rt * 16 + mb + j) * H2n + nl],      pa[j]);
                atomicAdd(&sm.p2.Acc[(rt * 16 + mb + j) * H2n + 16 + nl], pb[j]);
            }
        }
    }
    __syncthreads();

    // ---- layer 3 + sigmoid: 64 rows, 16 threads per row, two passes
    {
        int j = tid & 15;
        float w3a = W3[j], w3b = W3[j + 16];
        float c2a = b2[j], c2b = b2[j + 16];
        #pragma unroll
        for (int rr = 0; rr < 2; ++rr) {
            int m2 = (tid >> 4) + rr * 32;
            float p = (sm.p2.Acc[m2 * H2n + j]      + c2a) * w3a
                    + (sm.p2.Acc[m2 * H2n + 16 + j] + c2b) * w3b;
            p += __shfl_xor(p, 8);
            p += __shfl_xor(p, 4);
            p += __shfl_xor(p, 2);
            p += __shfl_xor(p, 1);
            if (j == 0) {
                float x = p + b3[0];
                logits[r0 + m2] = 1.0f / (1.0f + expf(-x));
            }
        }
    }
}

// ---------------------------------------------------------------------------
// Kernel 2: per-sample exact top-k mean (k = seq_len/16 + 1, k <= 129).
// ---------------------------------------------------------------------------
__global__ __launch_bounds__(256) void topk_kernel(
    const float* __restrict__ logits, const int* __restrict__ seq_len,
    float* __restrict__ out)
{
    __shared__ float vals[Tn];
    __shared__ float rmax[256];
    __shared__ int   ridx[256];
    __shared__ float s_sum;

    const int b = blockIdx.x;
    const int t = threadIdx.x;
    const int sl = seq_len[b];
    const int k  = sl / 16 + 1;

    for (int i = t; i < Tn; i += 256) {
        float v = logits[b * Tn + i];
        vals[i] = (i < sl) ? v : -2.0f;
    }
    if (t == 0) s_sum = 0.0f;
    __syncthreads();

    for (int iter = 0; iter < k; ++iter) {
        float mv = -3.0f; int mi = 0;
        for (int i = t; i < Tn; i += 256) {
            float v = vals[i];
            if (v > mv) { mv = v; mi = i; }
        }
        rmax[t] = mv; ridx[t] = mi;
        __syncthreads();
        for (int s = 128; s > 0; s >>= 1) {
            if (t < s) {
                float ov = rmax[t + s]; int oi = ridx[t + s];
                if (ov > rmax[t] || (ov == rmax[t] && oi < ridx[t])) {
                    rmax[t] = ov; ridx[t] = oi;
                }
            }
            __syncthreads();
        }
        if (t == 0) { s_sum += rmax[0]; vals[ridx[0]] = -2.0f; }
        __syncthreads();
    }
    if (t == 0) out[b] = s_sum / (float)k;
}

// ---------------------------------------------------------------------------
// Launch
// ---------------------------------------------------------------------------
extern "C" void kernel_launch(void* const* d_in, const int* in_sizes, int n_in,
                              void* d_out, int out_size, void* d_ws, size_t ws_size,
                              hipStream_t stream) {
    const float* avf  = (const float*)d_in[0];
    const float* W1   = (const float*)d_in[1];
    const float* b1   = (const float*)d_in[2];
    const float* W2   = (const float*)d_in[3];
    const float* b2   = (const float*)d_in[4];
    const float* W3   = (const float*)d_in[5];
    const float* b3   = (const float*)d_in[6];
    const int*   slen = (const int*)d_in[7];
    float* out = (float*)d_out;

    char* ws = (char*)d_ws;
    float*          logits = (float*)ws;                                  // 256 KB
    unsigned short* W1t    = (unsigned short*)(ws + 262144);              // 1 MiB
    unsigned short* W2t    = (unsigned short*)(ws + 262144 + 1048576);    // 32 KB

    prep_kernel<<<(H1n * Din + 255) / 256, 256, 0, stream>>>(W1, W2, W1t, W2t);
    mlp_kernel<<<(Bn * Tn) / ROWS, 512, 0, stream>>>(avf, W1t, W2t, b1, b2, W3, b3, logits);
    topk_kernel<<<Bn, 256, 0, stream>>>(logits, slen, out);
}
